// HeteroGNN_1099511628121
// MI455X (gfx1250) — compile-verified, matrix operand order checked by reading, not run on hardware
//
#include <hip/hip_runtime.h>
#include <hip/hip_bf16.h>

typedef _Float16 v4h  __attribute__((ext_vector_type(4)));
typedef _Float16 v8h  __attribute__((ext_vector_type(8)));
typedef _Float16 v16h __attribute__((ext_vector_type(16)));
typedef float    v8f  __attribute__((ext_vector_type(8)));

#define N_USER 100000
#define N_ITEM 200000
#define HDIM   128

#define WT_PITCH 136   // 128 (max K) + 8 halfs pad; 272B rows -> 16B aligned
#define AS_PITCH 40    // 32 + 8 halfs pad; 80B rows -> 16B aligned

// Fused GEMM: C[M,128] = op(A[M,K] @ W[K,128] (+bias) (+C) (relu?))
// 256 threads = 8 waves; block computes 256 rows x 128 cols (32 rows/wave).
// W held fully in LDS as f16 transposed [n][k]; A staged per 32-wide K chunk.
// Each B fragment feeds two WMMAs (two 16-row strips) -> half the LDS B traffic.
template<int ACC, int BIAS, int RELU>
__global__ __launch_bounds__(256)
void gemm_nk128(const float* __restrict__ A, const float* __restrict__ W,
                const float* __restrict__ bias, float* __restrict__ C,
                int M, int K) {
  __shared__ __align__(16) _Float16 Wt[128 * WT_PITCH];  // 34816 B
  __shared__ __align__(16) _Float16 As[256 * AS_PITCH];  // 20480 B

  const int tid  = threadIdx.x;
  const int row0 = blockIdx.x * 256;
  const int Mm1  = M - 1;

  // Stage W (fp32 [k][n]) -> LDS f16 [n][k], float4 global reads
  for (int idx = tid; idx < K * 32; idx += 256) {
    int k  = idx >> 5;
    int n4 = (idx & 31) << 2;
    const float4 v = *(const float4*)(W + (k << 7) + n4);
    Wt[(n4 + 0) * WT_PITCH + k] = (_Float16)v.x;
    Wt[(n4 + 1) * WT_PITCH + k] = (_Float16)v.y;
    Wt[(n4 + 2) * WT_PITCH + k] = (_Float16)v.z;
    Wt[(n4 + 3) * WT_PITCH + k] = (_Float16)v.w;
  }

  const int wave  = tid >> 5;
  const int lane  = tid & 31;
  const int mrow  = lane & 15;          // tile row (A) / tile col (B,D)
  const int kbase = (lane >> 4) * 8;    // K sub-base per WMMA 16-bit layout

  v8f acc0[8], acc1[8];
#pragma unroll
  for (int t = 0; t < 8; ++t) { acc0[t] = (v8f)0.f; acc1[t] = (v8f)0.f; }

  const int nchunks = K >> 5;
  for (int kc = 0; kc < nchunks; ++kc) {
    __syncthreads();
    // Stage A chunk [256 rows x 32 k] -> LDS f16; float4 reads, v4h (8B) stores.
    // Edge guard = row clamp: clamped rows only affect D rows >= M (never stored).
    for (int idx = tid; idx < 2048; idx += 256) {
      int r  = idx >> 3;
      int c4 = (idx & 7) << 2;
      int gr = min(row0 + r, Mm1);
      const float4 v = *(const float4*)(A + (size_t)gr * K + (kc * 32 + c4));
      v4h h;
      h[0] = (_Float16)v.x; h[1] = (_Float16)v.y;
      h[2] = (_Float16)v.z; h[3] = (_Float16)v.w;
      *(v4h*)(&As[r * AS_PITCH + c4]) = h;
    }
    __syncthreads();

    // Two A fragments: rows wave*32 + {0..15} and wave*32 + {16..31}
    const _Float16* ap0 = &As[(wave * 32 + mrow) * AS_PITCH + kbase];
    const _Float16* ap1 = ap0 + 16 * AS_PITCH;
    v8h a0lo = *(const v8h*)(ap0);
    v8h a0hi = *(const v8h*)(ap0 + 16);
    v8h a1lo = *(const v8h*)(ap1);
    v8h a1hi = *(const v8h*)(ap1 + 16);
    v16h a0, a1;
#pragma unroll
    for (int i = 0; i < 8; ++i) {
      a0[i] = a0lo[i]; a0[i + 8] = a0hi[i];
      a1[i] = a1lo[i]; a1[i + 8] = a1hi[i];
    }

#pragma unroll
    for (int nt = 0; nt < 8; ++nt) {
      const _Float16* bp = &Wt[(nt * 16 + mrow) * WT_PITCH + kc * 32 + kbase];
      v8h blo = *(const v8h*)(bp);
      v8h bhi = *(const v8h*)(bp + 16);
      v16h b;
#pragma unroll
      for (int i = 0; i < 8; ++i) { b[i] = blo[i]; b[i + 8] = bhi[i]; }
      acc0[nt] = __builtin_amdgcn_wmma_f32_16x16x32_f16(
          false, a0, false, b, (short)0, acc0[nt], false, false);
      acc1[nt] = __builtin_amdgcn_wmma_f32_16x16x32_f16(
          false, a1, false, b, (short)0, acc1[nt], false, false);
    }
  }

  // D layout: lane l, VGPR j -> row = j + 8*(l>>4), col = l&15 (per 16x16 tile)
  const int jrow  = (lane >> 4) * 8;
  const int base0 = row0 + wave * 32 + jrow;
#pragma unroll
  for (int nt = 0; nt < 8; ++nt) {
    const int col = nt * 16 + mrow;
    const float bv = BIAS ? bias[col] : 0.f;
#pragma unroll
    for (int j = 0; j < 8; ++j) {
      int gr = base0 + j;
      if (gr < M) {
        size_t off = (size_t)gr * 128 + col;
        float v = acc0[nt][j] + bv;
        if (ACC) v += C[off];
        if (RELU) v = fmaxf(v, 0.f);
        C[off] = v;
      }
    }
#pragma unroll
    for (int j = 0; j < 8; ++j) {
      int gr = base0 + 16 + j;
      if (gr < M) {
        size_t off = (size_t)gr * 128 + col;
        float v = acc1[nt][j] + bv;
        if (ACC) v += C[off];
        if (RELU) v = fmaxf(v, 0.f);
        C[off] = v;
      }
    }
  }
}

// One wave per edge: agg[dst[e]][:] += T[src[e]][:]  (128 floats, float4/lane)
__global__ __launch_bounds__(256)
void scatter_add(const float* __restrict__ T, const int* __restrict__ src,
                 const int* __restrict__ dst, float* __restrict__ agg, int E) {
  int e = blockIdx.x * 8 + (threadIdx.x >> 5);
  int lane = threadIdx.x & 31;
  if (e >= E) return;
  int s = src[e], d = dst[e];
  const float4 v = *(const float4*)(T + (size_t)s * 128 + lane * 4);
  float* p = agg + (size_t)d * 128 + lane * 4;
  unsafeAtomicAdd(p + 0, v.x);
  unsafeAtomicAdd(p + 1, v.y);
  unsafeAtomicAdd(p + 2, v.z);
  unsafeAtomicAdd(p + 3, v.w);
}

// One wave per row: out[r] = dot(t1[r,:], W2) + b2
__global__ __launch_bounds__(256)
void head_gemv(const float* __restrict__ t1, const float* __restrict__ W2,
               const float* __restrict__ b2, float* __restrict__ out, int M) {
  int row = blockIdx.x * 8 + (threadIdx.x >> 5);
  int lane = threadIdx.x & 31;
  if (row >= M) return;
  const float4 x = *(const float4*)(t1 + (size_t)row * 128 + lane * 4);
  const float4 w = *(const float4*)(W2 + lane * 4);
  float s = x.x * w.x + x.y * w.y + x.z * w.z + x.w * w.w;
#pragma unroll
  for (int off = 16; off > 0; off >>= 1) s += __shfl_xor(s, off, 32);
  if (lane == 0) out[row] = s + b2[0];
}

extern "C" void kernel_launch(void* const* d_in, const int* in_sizes, int n_in,
                              void* d_out, int out_size, void* d_ws, size_t ws_size,
                              hipStream_t stream) {
  const float* x_user = (const float*)d_in[0];
  const float* x_item = (const float*)d_in[1];
  const float* puW    = (const float*)d_in[2];
  const float* pub    = (const float*)d_in[3];
  const float* piW    = (const float*)d_in[4];
  const float* pib    = (const float*)d_in[5];
  const float* WsrcB  = (const float*)d_in[6];
  const float* WdstB  = (const float*)d_in[7];
  const float* bB     = (const float*)d_in[8];
  const float* WsrcR  = (const float*)d_in[9];
  const float* WdstR  = (const float*)d_in[10];
  const float* bR     = (const float*)d_in[11];
  const float* hW1    = (const float*)d_in[12];
  const float* hb1    = (const float*)d_in[13];
  const float* hW2    = (const float*)d_in[14];
  const float* hb2    = (const float*)d_in[15];
  const int* ebs = (const int*)d_in[16];
  const int* ebd = (const int*)d_in[17];
  const int* ers = (const int*)d_in[18];
  const int* erd = (const int*)d_in[19];
  const int E = in_sizes[16];
  float* out = (float*)d_out;

  char* ws = (char*)d_ws;
  const size_t SU = (size_t)N_USER * HDIM * sizeof(float);  // 51.2 MB
  const size_t SI = (size_t)N_ITEM * HDIM * sizeof(float);  // 102.4 MB
  float* hu = (float*)(ws);
  float* hi = (float*)(ws + SU);
  float* nu = (float*)(ws + SU + SI);
  float* ni = (float*)(ws + 2 * SU + SI);
  float* T  = (float*)(ws + 2 * SU + 2 * SI);               // max(N) x 128 temp

  const dim3 blk(256);
  const dim3 gU((N_USER + 255) / 256);
  const dim3 gI((N_ITEM + 255) / 256);
  const dim3 gE((E + 7) / 8);

  // Projections + ReLU
  gemm_nk128<0, 1, 1><<<gU, blk, 0, stream>>>(x_user, puW, pub, hu, N_USER, 128);
  gemm_nk128<0, 1, 1><<<gI, blk, 0, stream>>>(x_item, piW, pib, hi, N_ITEM, 64);

  for (int l = 0; l < 2; ++l) {
    const float* Wsb = WsrcB + (size_t)l * HDIM * HDIM;
    const float* Wdb = WdstB + (size_t)l * HDIM * HDIM;
    const float* bb  = bB    + (size_t)l * HDIM;
    const float* Wsr = WsrcR + (size_t)l * HDIM * HDIM;
    const float* Wdr = WdstR + (size_t)l * HDIM * HDIM;
    const float* br  = bR    + (size_t)l * HDIM;

    // buys: user -> item
    gemm_nk128<0, 0, 0><<<gU, blk, 0, stream>>>(hu, Wsb, nullptr, T, N_USER, 128);
    hipMemsetAsync(ni, 0, SI, stream);
    scatter_add<<<gE, blk, 0, stream>>>(T, ebs, ebd, ni, E);
    gemm_nk128<1, 1, 0><<<gI, blk, 0, stream>>>(hi, Wdb, bb, ni, N_ITEM, 128);

    // reviews: item -> user
    gemm_nk128<0, 0, 0><<<gI, blk, 0, stream>>>(hi, Wsr, nullptr, T, N_ITEM, 128);
    hipMemsetAsync(nu, 0, SU, stream);
    scatter_add<<<gE, blk, 0, stream>>>(T, ers, erd, nu, E);
    gemm_nk128<1, 1, 0><<<gU, blk, 0, stream>>>(hu, Wdr, br, nu, N_USER, 128);

    // swap double buffers
    float* t;
    t = hu; hu = nu; nu = t;
    t = hi; hi = ni; ni = t;
  }

  // Head: t1 = relu(hu @ W1 + b1); out = t1 @ W2 + b2
  gemm_nk128<0, 1, 1><<<gU, blk, 0, stream>>>(hu, hW1, hb1, T, N_USER, 128);
  head_gemv<<<dim3((N_USER + 7) / 8), blk, 0, stream>>>(T, hW2, hb2, out, N_USER);
}
